// RGLinear_6751688589356
// MI455X (gfx1250) — matrix operand order repaired and checked by reading, not compiled, not run on hardware
//
#include <hip/hip_runtime.h>
#include <hip/hip_bf16.h>

// ---------------------------------------------------------------------------
// Two-level VQ linear:  out = x @ dequant(W)^T + bias
// Pass 1: dequantize W -> bf16 workspace (32 MB).
// Pass 2: convert x -> bf16 workspace (64 MB).  (both L2-resident: 96 < 192 MB)
// Pass 3: bf16 WMMA GEMM, f32 accumulate, fused bias.
//         Tiles staged with CDNA5 async global->LDS DMA (ASYNCcnt). K-loop is
//         NOT unrolled so accumulators stay pinned for WMMA's in-place D=C
//         update; A fragments are software-rotated through 2 register slots
//         so each ds_load's latency hides behind the previous WMMA pair.
// ---------------------------------------------------------------------------

typedef __attribute__((ext_vector_type(16))) __bf16 v16bf;
typedef __attribute__((ext_vector_type(8)))  float  v8f;

#define IN_F   4096
#define OUT_F  4096
#define BLK    16
#define NBLOCKS (OUT_F * IN_F / BLK)   // 1,048,576

#define BM  128
#define BN  128
#define BK  32
#define BKP 40                // padded LDS row stride (elems): 80B rows, 16B-aligned,
                              // 20-bank stride -> conflict-free b128 lane pattern
#define NK  (IN_F / BK)       // 128 K-steps

// float -> bf16, round-to-nearest-even
__device__ __forceinline__ unsigned short f2bf(float x) {
  unsigned u = __float_as_uint(x);
  u += 0x7FFFu + ((u >> 16) & 1u);
  return (unsigned short)(u >> 16);
}

// ---------------------------------------------------------------------------
// Kernel 1: W_bf16[b*16+j] = coarse[cl[b]][j] + res[rl[b]][j]
// ---------------------------------------------------------------------------
__global__ __launch_bounds__(256) void vq_dequant_bf16(
    const float* __restrict__ cc, const int* __restrict__ cl,
    const float* __restrict__ rc, const int* __restrict__ rl,
    unsigned short* __restrict__ Wb) {
  int b = blockIdx.x * blockDim.x + threadIdx.x;
  if (b >= NBLOCKS) return;
  const float4* c4 = (const float4*)(cc + (size_t)cl[b] * BLK);
  const float4* r4 = (const float4*)(rc + (size_t)rl[b] * BLK);
  union { unsigned short s[16]; uint4 q[2]; } pk;
#pragma unroll
  for (int v = 0; v < 4; ++v) {
    float4 a = c4[v], d = r4[v];
    pk.s[4 * v + 0] = f2bf(a.x + d.x);
    pk.s[4 * v + 1] = f2bf(a.y + d.y);
    pk.s[4 * v + 2] = f2bf(a.z + d.z);
    pk.s[4 * v + 3] = f2bf(a.w + d.w);
  }
  uint4* o = (uint4*)(Wb + (size_t)b * BLK);
  o[0] = pk.q[0];
  o[1] = pk.q[1];
}

// ---------------------------------------------------------------------------
// Kernel 2: x (f32) -> bf16, 16 elems/thread
// ---------------------------------------------------------------------------
__global__ __launch_bounds__(256) void x_to_bf16(
    const float* __restrict__ X, unsigned short* __restrict__ Xb, int n16) {
  int i = blockIdx.x * blockDim.x + threadIdx.x;
  if (i >= n16) return;
  const float4* p = (const float4*)(X + (size_t)i * 16);
  union { unsigned short s[16]; uint4 q[2]; } pk;
#pragma unroll
  for (int v = 0; v < 4; ++v) {
    float4 a = p[v];
    pk.s[4 * v + 0] = f2bf(a.x);
    pk.s[4 * v + 1] = f2bf(a.y);
    pk.s[4 * v + 2] = f2bf(a.z);
    pk.s[4 * v + 3] = f2bf(a.w);
  }
  uint4* o = (uint4*)(Xb + (size_t)i * 16);
  o[0] = pk.q[0];
  o[1] = pk.q[1];
}

// ---------------------------------------------------------------------------
// Kernel 3: C[M,N] = Xb @ Wb^T + bias.  128x128x32 tiles, double-buffered LDS
// filled by global_load_async_to_lds_b128; 8 wave32 waves (2M x 4N), 8 WMMA
// per wave per K-step.
// ---------------------------------------------------------------------------
__global__ __launch_bounds__(256) void vq_gemm_bf16(
    const unsigned short* __restrict__ Xb, const unsigned short* __restrict__ Wb,
    const float* __restrict__ bias, float* __restrict__ C) {
  __shared__ __align__(16) unsigned short lA[2][BM * BKP];  // 2 x 10,240 B
  __shared__ __align__(16) unsigned short lB[2][BN * BKP];  // 2 x 10,240 B

  const int tid  = threadIdx.x;
  const int lane = tid & 31;
  const int wave = tid >> 5;
  const int l15  = lane & 15;
  const int lh   = lane >> 4;
  const int wm   = (wave & 1) * 64;   // wave M offset in tile (4 frags)
  const int wn   = (wave >> 1) * 32;  // wave N offset in tile (2 frags)

  const int bm = blockIdx.y * BM;
  const int bn = blockIdx.x * BN;

  // staging: thread t owns row t/2, 16-col half (t&1) of both tiles (32B each)
  const int srow = tid >> 1;
  const int scol = (tid & 1) * 16;
  const unsigned gOffA = (unsigned)(((bm + srow) * IN_F + scol) * 2);  // bytes
  const unsigned gOffB = (unsigned)(((bn + srow) * IN_F + scol) * 2);  // bytes
  // generic->LDS: hardware truncates flat addr to addr[31:0] for LDS aperture
  const unsigned ldsA[2] = {
      (unsigned)(uintptr_t)&lA[0][srow * BKP + scol],
      (unsigned)(uintptr_t)&lA[1][srow * BKP + scol]};
  const unsigned ldsB[2] = {
      (unsigned)(uintptr_t)&lB[0][srow * BKP + scol],
      (unsigned)(uintptr_t)&lB[1][srow * BKP + scol]};

  auto stage = [&](int buf, int k0) {
    const unsigned va = gOffA + (unsigned)(k0 * 2);
    const unsigned vb = gOffB + (unsigned)(k0 * 2);
    // offset: applies to BOTH global and LDS addresses (ISA async pseudocode)
    asm volatile("global_load_async_to_lds_b128 %0, %1, %2 offset:0"
                 :: "v"(ldsA[buf]), "v"(va), "s"(Xb));
    asm volatile("global_load_async_to_lds_b128 %0, %1, %2 offset:16"
                 :: "v"(ldsA[buf]), "v"(va), "s"(Xb));
    asm volatile("global_load_async_to_lds_b128 %0, %1, %2 offset:0"
                 :: "v"(ldsB[buf]), "v"(vb), "s"(Wb));
    asm volatile("global_load_async_to_lds_b128 %0, %1, %2 offset:16"
                 :: "v"(ldsB[buf]), "v"(vb), "s"(Wb));
  };

  v8f acc[4][2];
#pragma unroll
  for (int i = 0; i < 4; ++i)
#pragma unroll
    for (int j = 0; j < 2; ++j) acc[i][j] = (v8f)(0.0f);

  union Frag { uint4 q[2]; v16bf v; };

  // B frags once per K-step; A frags rotate through 2 slots so the ds_load of
  // frag i+1 issues before the WMMAs of frag i (hides LDS latency).
  auto compute = [&](int cur) {
    const unsigned short* As = &lA[cur][0];
    const unsigned short* Bs = &lB[cur][0];
    // B frags: 32x16 bf16 (B col n = W row n). Lane l: col n0+(l&15); K runs {8h..8h+7}, {16+8h..}
    Frag bfr[2];
#pragma unroll
    for (int j = 0; j < 2; ++j) {
      const int ro = (wn + 16 * j + l15) * BKP + 8 * lh;
      bfr[j].q[0] = *(const uint4*)(Bs + ro);
      bfr[j].q[1] = *(const uint4*)(Bs + ro + 16);
    }
    // A frags: 16x32 bf16. Lane l: row m0+(l&15); K runs {8h..8h+7}, {16+8h..}
    auto loadA = [&](Frag& f, int i) {
      const int ro = (wm + 16 * i + l15) * BKP;
      f.q[0] = *(const uint4*)(As + ro + 8 * lh);
      f.q[1] = *(const uint4*)(As + ro + 16 + 8 * lh);
    };
    Frag afr[2];
    loadA(afr[0], 0);
#pragma unroll
    for (int i = 0; i < 4; ++i) {
      if (i < 3) loadA(afr[(i + 1) & 1], i + 1);
      acc[i][0] = __builtin_amdgcn_wmma_f32_16x16x32_bf16(
          false, afr[i & 1].v, false, bfr[0].v, (short)0, acc[i][0], false, false);
      acc[i][1] = __builtin_amdgcn_wmma_f32_16x16x32_bf16(
          false, afr[i & 1].v, false, bfr[1].v, (short)0, acc[i][1], false, false);
    }
  };

  stage(0, 0);
  asm volatile("s_wait_asynccnt 0x0" ::: "memory");
  __syncthreads();

  // Uniform body, final K-step peeled.  NOT unrolled: with a single body only
  // one tile of fragments is live next to the 64 acc VGPRs, so the allocator
  // keeps accumulators stationary (in-place WMMA D=C, no copies, no nops).
  int cur = 0;
#pragma clang loop unroll(disable)
  for (int kt = 0; kt < NK - 1; ++kt) {
    stage(cur ^ 1, (kt + 1) * BK);
    compute(cur);
    asm volatile("s_wait_asynccnt 0x0" ::: "memory");
    __syncthreads();
    cur ^= 1;
  }
  compute(cur);

  // Epilogue: lane l holds n = n0+(l&15); VGPR r -> m = m0 + r + 8*(l>>4)
#pragma unroll
  for (int j = 0; j < 2; ++j) {
    const int n   = bn + wn + 16 * j + l15;
    const float bv = bias[n];
#pragma unroll
    for (int i = 0; i < 4; ++i) {
      const int m0 = bm + wm + 16 * i + 8 * lh;
#pragma unroll
      for (int r = 0; r < 8; ++r)
        __builtin_nontemporal_store(acc[i][j][r] + bv,
                                    &C[(size_t)(m0 + r) * OUT_F + n]);
    }
  }
}

// ---------------------------------------------------------------------------
extern "C" void kernel_launch(void* const* d_in, const int* in_sizes, int n_in,
                              void* d_out, int out_size, void* d_ws, size_t ws_size,
                              hipStream_t stream) {
  const float* x    = (const float*)d_in[0];
  const float* cc   = (const float*)d_in[1];
  const int*   cl   = (const int*)  d_in[2];
  const float* rc   = (const float*)d_in[3];
  const int*   rl   = (const int*)  d_in[4];
  const float* bias = (const float*)d_in[5];
  float*       out  = (float*)d_out;

  unsigned short* Wb = (unsigned short*)d_ws;            // 32 MB
  unsigned short* Xb = Wb + (size_t)OUT_F * IN_F;        // 64 MB

  const int M = in_sizes[0] / IN_F;                      // 8192

  vq_dequant_bf16<<<NBLOCKS / 256, 256, 0, stream>>>(cc, cl, rc, rl, Wb);

  const int n16 = M * IN_F / 16;
  x_to_bf16<<<n16 / 256, 256, 0, stream>>>(x, Xb, n16);

  dim3 grid(OUT_F / BN, M / BM);                         // (32, 64)
  vq_gemm_bf16<<<grid, 256, 0, stream>>>(Xb, Wb, bias, out);
}
